// YOLOv3_22608707846975
// MI455X (gfx1250) — compile-verified, matrix operand order checked by reading, not run on hardware
//
#include <hip/hip_runtime.h>
#include <stdint.h>
#include <stddef.h>

// ---------------------------------------------------------------------------
// Types
// ---------------------------------------------------------------------------
typedef unsigned short bf16_t;  // raw bfloat16 storage
typedef __attribute__((ext_vector_type(16))) __bf16 v16bf;
typedef __attribute__((ext_vector_type(8)))  float  v8f;
typedef __attribute__((ext_vector_type(4)))  unsigned int u32x4;
typedef __attribute__((ext_vector_type(8)))  int    i32x8;
typedef __attribute__((ext_vector_type(4)))  int    i32x4;

#define BN_EPS 1e-3f

#if __has_builtin(__builtin_amdgcn_tensor_load_to_lds)
#define HAVE_TDM 1
#else
#define HAVE_TDM 0
#endif

__device__ __forceinline__ float bf2f(bf16_t v) {
  unsigned u = ((unsigned)v) << 16; float f; __builtin_memcpy(&f, &u, 4); return f;
}
__device__ __forceinline__ bf16_t f2bf(float f) {
  unsigned u; __builtin_memcpy(&u, &f, 4);
  unsigned r = (u + 0x7fffu + ((u >> 16) & 1u)) >> 16;  // round-to-nearest-even
  return (bf16_t)r;
}
__device__ __forceinline__ float sigmoidf_(float x) { return 1.0f / (1.0f + __expf(-x)); }
__device__ __forceinline__ int imin_(int a, int b) { return a < b ? a : b; }

// ---------------------------------------------------------------------------
// Weight repack: f32 HWIO -> bf16 [Cout][K], K = ky*kx*ci contiguous
// src flat index of (k, co) is k*Cout + co.
// ---------------------------------------------------------------------------
struct PackP { const float* w; bf16_t* wp; int K; int Cout; };
__global__ void k_pack_w(PackP p) {
  long long i = (long long)blockIdx.x * 256 + threadIdx.x;
  long long tot = (long long)p.K * p.Cout;
  if (i >= tot) return;
  int co = (int)(i / p.K);
  int k  = (int)(i % p.K);
  p.wp[i] = f2bf(p.w[(long long)k * p.Cout + co]);
}

// BN fold: sb[0..C) = scale, sb[C..2C) = bias
struct FuseP { const float *g, *b, *m, *v; float* sb; int C; };
__global__ void k_fuse_bn(FuseP p) {
  int c = blockIdx.x * 256 + threadIdx.x;
  if (c >= p.C) return;
  float s = p.g[c] * rsqrtf(p.v[c] + BN_EPS);
  p.sb[c] = s;
  p.sb[p.C + c] = p.b[c] - p.m[c] * s;
}

// ---------------------------------------------------------------------------
// TDM: stage a [32 rows x len k] bf16 tile (row stride K elems) into LDS
// ---------------------------------------------------------------------------
__device__ __forceinline__ void tdm_stage(const bf16_t* g, unsigned ldsOff,
                                          int len, int K, int Cout) {
#if HAVE_TDM
  unsigned long long ga = (unsigned long long)(uintptr_t)g;
  u32x4 g0 = {0u, 0u, 0u, 0u};
  g0[0] = 1u;                                              // count = 1 (valid D#)
  g0[1] = ldsOff;                                          // lds_addr
  g0[2] = (unsigned)(ga & 0xffffffffull);                  // global_addr[31:0]
  g0[3] = (unsigned)((ga >> 32) & 0x01ffffffull) | (2u << 30);  // addr[56:32]|type=2
  const unsigned td0 = (unsigned)K;
  const unsigned td1 = (unsigned)Cout;
  i32x8 g1 = {0, 0, 0, 0, 0, 0, 0, 0};
  g1[0] = (int)(1u << 16);                                 // data_size = 2 bytes
  g1[1] = (int)((td0 & 0xffffu) << 16);                    // tensor_dim0 lo
  g1[2] = (int)(((td0 >> 16) & 0xffffu) | ((td1 & 0xffffu) << 16));
  g1[3] = (int)(((td1 >> 16) & 0xffffu) | (((unsigned)len) << 16)); // tile_dim0=len
  g1[4] = (int)32u;                                        // tile_dim1 = 32 rows
  g1[5] = (int)td0;                                        // tensor_dim0_stride lo32
  i32x4 gz4 = {0, 0, 0, 0};
#if __clang_major__ >= 23
  i32x8 gz8 = {0, 0, 0, 0, 0, 0, 0, 0};
  __builtin_amdgcn_tensor_load_to_lds(g0, g1, gz4, gz4, gz8, 0);
#else
  __builtin_amdgcn_tensor_load_to_lds(g0, g1, gz4, gz4, 0);
#endif
#else
  (void)g; (void)ldsOff; (void)len; (void)K; (void)Cout;
#endif
}

// ---------------------------------------------------------------------------
// Implicit-GEMM conv + BN + leaky (+ residual) via WMMA bf16.
// Block = 256 thr = 8 waves as 4(M) x 2(N); wave tile = 32(M pix) x 32(N cout).
// Weights stream through LDS in 32x128 chunks via the Tensor Data Mover,
// double-buffered (prefetch chunk c+1 while computing chunk c).
// ---------------------------------------------------------------------------
struct ConvP {
  const bf16_t* in; const bf16_t* wp; const float* sb; const bf16_t* res; bf16_t* out;
  int B, Hin, Win, Cin, Hout, Wout, Cout, ks, stride, pad, K;
};

__global__ __launch_bounds__(256) void k_conv_wmma(ConvP p) {
#if HAVE_TDM
  __shared__ __align__(16) bf16_t ldsW[2 * 2 * 32 * 128];  // [nw][buf][32r x 128k] = 32KB
#endif
  const int tid  = threadIdx.x;
  const int wave = tid >> 5;
  const int lane = tid & 31;
  const int lm = lane & 15;
  const int lh = lane >> 4;
  const int mw = wave & 3;       // 4 M-waves
  const int nw = wave >> 2;      // 2 N-waves
  const int HW = p.Hout * p.Wout;
  const int Npix = p.B * HW;
  const int m0 = blockIdx.x * 128 + mw * 32;
  int nt = blockIdx.y * 64 + nw * 32;
  if (nt + 32 > p.Cout) nt = p.Cout - 32;   // Cout is always a multiple of 32

  // per-lane pixel decomposition for the wave's two M subtiles
  int bi[2], oy[2], ox[2]; bool mok[2];
#pragma unroll
  for (int s = 0; s < 2; ++s) {
    int m = m0 + s * 16 + lm;
    mok[s] = m < Npix;
    int mm = mok[s] ? m : 0;
    bi[s] = mm / HW; int r = mm % HW; oy[s] = r / p.Wout; ox[s] = r % p.Wout;
  }

#if HAVE_TDM
  bf16_t* tile[2] = { &ldsW[(nw * 2 + 0) * 32 * 128], &ldsW[(nw * 2 + 1) * 32 * 128] };
  const unsigned ldsOff[2] = { (unsigned)(uintptr_t)(void*)tile[0],
                               (unsigned)(uintptr_t)(void*)tile[1] };
#endif

  v8f acc[2][2];
#pragma unroll
  for (int s = 0; s < 2; ++s)
#pragma unroll
    for (int t2 = 0; t2 < 2; ++t2)
      acc[s][t2] = (v8f){0.f, 0.f, 0.f, 0.f, 0.f, 0.f, 0.f, 0.f};

  const u32x4 zero4 = {0u, 0u, 0u, 0u};
  const int off0 = lh ? 8 : 0;     // 16-bit A/B fragment K layout (wave32)
  const int off1 = lh ? 24 : 16;

  // division-free implicit-GEMM bookkeeping
  int cb = 0, kx = 0, ky = 0;
  const bf16_t* ap[2]; bool av[2];
  auto setA = [&]() {
#pragma unroll
    for (int s = 0; s < 2; ++s) {
      int iy = oy[s] * p.stride + ky - p.pad;
      int ix = ox[s] * p.stride + kx - p.pad;
      av[s] = mok[s] && iy >= 0 && iy < p.Hin && ix >= 0 && ix < p.Win;
      ap[s] = p.in + (((size_t)bi[s] * p.Hin + iy) * p.Win + ix) * p.Cin;
    }
  };
  setA();

  const int Kc = 128;
  const int nch = (p.K + Kc - 1) / Kc;
  const size_t wrow = (size_t)nt * p.K;

#if HAVE_TDM
  if (mw == 0) tdm_stage(p.wp + wrow, ldsOff[0], imin_(Kc, p.K), p.K, p.Cout);
#endif

  for (int c = 0; c < nch; ++c) {
    const int kc = c * Kc;
    const int len = imin_(Kc, p.K - kc);
#if HAVE_TDM
    if (mw == 0) {
      if (c + 1 < nch) {
        tdm_stage(p.wp + wrow + kc + Kc, ldsOff[(c + 1) & 1],
                  imin_(Kc, p.K - (kc + Kc)), p.K, p.Cout);
        __builtin_amdgcn_s_wait_tensorcnt((short)1);  // chunk c landed
      } else {
        __builtin_amdgcn_s_wait_tensorcnt((short)0);  // drain
      }
    }
    __syncthreads();   // publish LDS tile to all waves
    const bf16_t* tb = tile[c & 1];
#endif
#pragma unroll
    for (int j = 0; j < 4; ++j) {
      if (j * 32 < len) {
        const int koff = j * 32;
        // A fragments: two aligned 16B global loads per M subtile
        union { u32x4 u[2]; v16bf v; } A[2];
#pragma unroll
        for (int s = 0; s < 2; ++s) {
          A[s].u[0] = zero4; A[s].u[1] = zero4;
          if (av[s]) {
            const bf16_t* a = ap[s] + cb;
            A[s].u[0] = *(const u32x4*)(a + off0);
            A[s].u[1] = *(const u32x4*)(a + off1);
          }
        }
        // B fragments: two 16B reads per N subtile
        union { u32x4 u[2]; v16bf v; } Bf[2];
#pragma unroll
        for (int t2 = 0; t2 < 2; ++t2) {
#if HAVE_TDM
          const bf16_t* b = tb + (size_t)(t2 * 16 + lm) * len + koff;
#else
          const bf16_t* b = p.wp + (size_t)(nt + t2 * 16 + lm) * p.K + kc + koff;
#endif
          Bf[t2].u[0] = *(const u32x4*)(b + off0);
          Bf[t2].u[1] = *(const u32x4*)(b + off1);
        }
#pragma unroll
        for (int s = 0; s < 2; ++s)
#pragma unroll
          for (int t2 = 0; t2 < 2; ++t2)
            acc[s][t2] = __builtin_amdgcn_wmma_f32_16x16x32_bf16(
                false, A[s].v, false, Bf[t2].v, (short)0, acc[s][t2], false, false);
        // advance (ky,kx,ci) without division
        cb += 32;
        if (cb == p.Cin) {
          cb = 0; ++kx;
          if (kx == p.ks) { kx = 0; ++ky; }
          setA();
        }
      }
    }
#if HAVE_TDM
    __syncthreads();   // all waves done with tile[c&1] before it is re-staged
#endif
  }

  // ---- epilogue: BN + leaky + optional residual, store bf16 NHWC
#pragma unroll
  for (int t2 = 0; t2 < 2; ++t2) {
    const int n = nt + t2 * 16 + lm;
    const float scl = p.sb[n];
    const float bia = p.sb[p.Cout + n];
#pragma unroll
    for (int s = 0; s < 2; ++s) {
#pragma unroll
      for (int r = 0; r < 8; ++r) {
        const int mr = m0 + s * 16 + (lh ? 8 + r : r);
        if (mr >= Npix) continue;
        const int bi2 = mr / HW;
        const int rr  = mr % HW;
        const int oy2 = rr / p.Wout, ox2 = rr % p.Wout;
        const size_t oi = (((size_t)bi2 * p.Hout + oy2) * p.Wout + ox2) * p.Cout + n;
        float v = acc[s][t2][r] * scl + bia;
        v = v > 0.f ? v : 0.1f * v;
        if (p.res) v += bf2f(p.res[oi]);
        p.out[oi] = f2bf(v);
      }
    }
  }
}

// ---------------------------------------------------------------------------
// First conv: 3x3, 3->32, stride 1, SAME, BN + leaky; f32 input -> bf16 out
// ---------------------------------------------------------------------------
struct C0P { const float* x; const float *w, *g, *b, *m, *v; bf16_t* out; };
__global__ void k_conv0(C0P p) {
  long long i = (long long)blockIdx.x * 256 + threadIdx.x;
  const long long tot = 2LL * 416 * 416 * 32;
  if (i >= tot) return;
  int co = (int)(i & 31);
  long long pix = i >> 5;
  int ox = (int)(pix % 416); long long t = pix / 416;
  int oy = (int)(t % 416);   int bi = (int)(t / 416);
  float acc = 0.f;
  for (int ky = 0; ky < 3; ++ky) {
    int iy = oy + ky - 1; if (iy < 0 || iy >= 416) continue;
    for (int kx = 0; kx < 3; ++kx) {
      int ix = ox + kx - 1; if (ix < 0 || ix >= 416) continue;
      const float* xp = p.x + (((long long)bi * 416 + iy) * 416 + ix) * 3;
      const float* wp = p.w + ((ky * 3 + kx) * 3) * 32 + co;
      acc += xp[0] * wp[0] + xp[1] * wp[32] + xp[2] * wp[64];
    }
  }
  float s = p.g[co] * rsqrtf(p.v[co] + BN_EPS);
  float v2 = (acc - p.m[co]) * s + p.b[co];
  v2 = v2 > 0.f ? v2 : 0.1f * v2;
  p.out[i] = f2bf(v2);
}

// ---------------------------------------------------------------------------
// Upsample x2 into (possibly wider) channel-strided buffer
// ---------------------------------------------------------------------------
struct UpP { const bf16_t* in; bf16_t* out; int B, H, W, C, Cs, off; };
__global__ void k_upsample(UpP p) {
  long long i = (long long)blockIdx.x * 256 + threadIdx.x;
  long long tot = (long long)p.B * p.H * 2 * p.W * 2 * p.C;
  if (i >= tot) return;
  int c = (int)(i % p.C); long long t = i / p.C;
  int x2 = (int)(t % (p.W * 2)); t /= (p.W * 2);
  int y2 = (int)(t % (p.H * 2)); int bi = (int)(t / (p.H * 2));
  bf16_t v = p.in[(((size_t)bi * p.H + y2 / 2) * p.W + x2 / 2) * p.C + c];
  p.out[(((size_t)bi * p.H * 2 + y2) * p.W * 2 + x2) * p.Cs + p.off + c] = v;
}

struct CpP { const bf16_t* in; bf16_t* out; int B, H, W, C, Cs, off; };
__global__ void k_copych(CpP p) {
  long long i = (long long)blockIdx.x * 256 + threadIdx.x;
  long long tot = (long long)p.B * p.H * p.W * p.C;
  if (i >= tot) return;
  int c = (int)(i % p.C);
  long long pix = i / p.C;
  p.out[(size_t)pix * p.Cs + p.off + c] = p.in[i];
}

// ---------------------------------------------------------------------------
// Head conv: 1x1 Cin->255 + bias, f32 output
// ---------------------------------------------------------------------------
struct HdP { const bf16_t* in; const float* w; const float* b; float* out; int Npix, Cin; };
__global__ void k_head(HdP p) {
  long long i = (long long)blockIdx.x * 256 + threadIdx.x;
  long long tot = (long long)p.Npix * 255;
  if (i >= tot) return;
  int co = (int)(i % 255);
  long long pix = i / 255;
  const bf16_t* ip = p.in + (size_t)pix * p.Cin;
  float acc = p.b[co];
  for (int ci = 0; ci < p.Cin; ++ci)
    acc += bf2f(ip[ci]) * p.w[(size_t)ci * 255 + co];
  p.out[i] = acc;
}

// ---------------------------------------------------------------------------
// YOLO decode -> bbox[B,10647,4], scores[B,10647,80]
// ---------------------------------------------------------------------------
struct DecP { const float* head; const float* anchors; float* bbox; float* scores;
              int B, G, rowoff, abase; };
__global__ void k_decode(DecP p) {
  long long i = (long long)blockIdx.x * 256 + threadIdx.x;
  long long tot = (long long)p.B * p.G * p.G * 3;
  if (i >= tot) return;
  int a = (int)(i % 3); long long t = i / 3;
  int gx = (int)(t % p.G); t /= p.G;
  int gy = (int)(t % p.G); int bi = (int)(t / p.G);
  const float* pr = p.head + (((size_t)bi * p.G + gy) * p.G + gx) * 255 + a * 85;
  float sx = sigmoidf_(pr[0]), sy = sigmoidf_(pr[1]);
  float bw = __expf(pr[2]) * p.anchors[(p.abase + a) * 2 + 0];
  float bh = __expf(pr[3]) * p.anchors[(p.abase + a) * 2 + 1];
  float cx = (sx + gx) / p.G, cy = (sy + gy) / p.G;
  size_t row = (size_t)p.rowoff + ((size_t)gy * p.G + gx) * 3 + a;
  float* bb = p.bbox + ((size_t)bi * 10647 + row) * 4;
  bb[0] = cx - bw * 0.5f; bb[1] = cy - bh * 0.5f;
  bb[2] = cx + bw * 0.5f; bb[3] = cy + bh * 0.5f;
  float obj = sigmoidf_(pr[4]);
  float* sc = p.scores + ((size_t)bi * 10647 + row) * 80;
  for (int c = 0; c < 80; ++c) sc[c] = obj * sigmoidf_(pr[5 + c]);
}

// ---------------------------------------------------------------------------
// NMS stage 1: per (image,class) top-256 (desc score, asc idx) + IoU suppress
// ---------------------------------------------------------------------------
struct N1P { const float* bbox; const float* scores; float* obx; float* osc; };
__global__ __launch_bounds__(256) void k_nms1(N1P p) {
  int b = blockIdx.x / 80, c = blockIdx.x % 80;
  const float* sc = p.scores + (size_t)b * 10647 * 80 + c;
  const float* bx = p.bbox + (size_t)b * 10647 * 4;
  __shared__ float rv[256]; __shared__ int ri[256];
  __shared__ float sel_s[256]; __shared__ int sel_i[256];
  int t = threadIdx.x;
  float prev_s = 3.4e38f; int prev_i = -1;
  for (int it = 0; it < 256; ++it) {
    float bv = -3.4e38f; int bi2 = 0x7fffffff;
    for (int i = t; i < 10647; i += 256) {
      float s = sc[(size_t)i * 80];
      bool lt = (s < prev_s) || (s == prev_s && i > prev_i);
      if (lt && (s > bv || (s == bv && i < bi2))) { bv = s; bi2 = i; }
    }
    rv[t] = bv; ri[t] = bi2; __syncthreads();
    for (int off = 128; off > 0; off >>= 1) {
      if (t < off) {
        float v2 = rv[t + off]; int i2 = ri[t + off];
        if (v2 > rv[t] || (v2 == rv[t] && i2 < ri[t])) { rv[t] = v2; ri[t] = i2; }
      }
      __syncthreads();
    }
    prev_s = rv[0]; prev_i = ri[0];
    if (t == 0) { sel_s[it] = prev_s; sel_i[it] = prev_i; }
    __syncthreads();
  }
  __shared__ float X1[256], Y1[256], X2[256], Y2[256], AR[256];
  __shared__ int keep[256];
  int gi = sel_i[t];
  float x1 = bx[(size_t)gi * 4 + 0], y1 = bx[(size_t)gi * 4 + 1];
  float x2 = bx[(size_t)gi * 4 + 2], y2 = bx[(size_t)gi * 4 + 3];
  X1[t] = x1; Y1[t] = y1; X2[t] = x2; Y2[t] = y2;
  AR[t] = fmaxf(x2 - x1, 0.f) * fmaxf(y2 - y1, 0.f);
  keep[t] = sel_s[t] > 0.5f ? 1 : 0;
  __syncthreads();
  for (int i = 0; i < 256; ++i) {
    if (keep[i] && t > i) {
      float lx = fmaxf(X1[i], x1), ly = fmaxf(Y1[i], y1);
      float rx = fminf(X2[i], x2), ry = fminf(Y2[i], y2);
      float inter = fmaxf(rx - lx, 0.f) * fmaxf(ry - ly, 0.f);
      float iou = inter / (AR[i] + AR[t] - inter + 1e-9f);
      if (iou > 0.5f) keep[t] = 0;
    }
    __syncthreads();
  }
  size_t o = (size_t)(b * 80 + c) * 256 + t;
  p.osc[o] = keep[t] ? sel_s[t] : -1.0f;
  p.obx[o * 4 + 0] = x1; p.obx[o * 4 + 1] = y1;
  p.obx[o * 4 + 2] = x2; p.obx[o * 4 + 3] = y2;
}

// ---------------------------------------------------------------------------
// NMS stage 2: per-image global top-100 + output packing
// d_out: boxes[2][100][4] | scores[2][100] | cls[2][100] (int) | count[2] (int)
// ---------------------------------------------------------------------------
struct N2P { const float* obx; const float* osc; float* out; };
__global__ __launch_bounds__(256) void k_nms2(N2P p) {
  int b = blockIdx.x;
  const float* sc = p.osc + (size_t)b * 80 * 256;
  const float* bx = p.obx + (size_t)b * 80 * 256 * 4;
  __shared__ float rv[256]; __shared__ int ri[256];
  __shared__ float ts[100]; __shared__ int ti[100];
  int t = threadIdx.x;
  float prev_s = 3.4e38f; int prev_i = -1;
  for (int it = 0; it < 100; ++it) {
    float bv = -3.4e38f; int bi2 = 0x7fffffff;
    for (int i = t; i < 20480; i += 256) {
      float s = sc[i];
      bool lt = (s < prev_s) || (s == prev_s && i > prev_i);
      if (lt && (s > bv || (s == bv && i < bi2))) { bv = s; bi2 = i; }
    }
    rv[t] = bv; ri[t] = bi2; __syncthreads();
    for (int off = 128; off > 0; off >>= 1) {
      if (t < off) {
        float v2 = rv[t + off]; int i2 = ri[t + off];
        if (v2 > rv[t] || (v2 == rv[t] && i2 < ri[t])) { rv[t] = v2; ri[t] = i2; }
      }
      __syncthreads();
    }
    prev_s = rv[0]; prev_i = ri[0];
    if (t == 0) { ts[it] = prev_s; ti[it] = prev_i; }
    __syncthreads();
  }
  float* obox = p.out;
  float* oscr = p.out + 800;
  int*   ocls = (int*)(p.out + 1000);
  int*   ocnt = (int*)(p.out + 1200);
  if (t < 100) {
    float s = ts[t]; int i = ti[t];
    int ok = s > 0.f;
    for (int k = 0; k < 4; ++k)
      obox[((size_t)b * 100 + t) * 4 + k] = ok ? bx[(size_t)i * 4 + k] : 0.f;
    oscr[b * 100 + t] = ok ? s : 0.f;
    ocls[b * 100 + t] = ok ? (i / 256) : 0;
  }
  if (t == 0) {
    int cnt = 0;
    for (int k = 0; k < 100; ++k) cnt += ts[k] > 0.f ? 1 : 0;
    ocnt[b] = cnt;
  }
}

// ---------------------------------------------------------------------------
// Host orchestration
// ---------------------------------------------------------------------------
extern "C" void kernel_launch(void* const* d_in, const int* in_sizes, int n_in,
                              void* d_out, int out_size, void* d_ws, size_t ws_size,
                              hipStream_t stream) {
  (void)in_sizes; (void)out_size; (void)ws_size;
  char* wsBase = (char*)d_ws;
  size_t wsOff = 0;
  auto alloc = [&](size_t bytes) -> void* {
    size_t a = (wsOff + 255) & ~(size_t)255;
    wsOff = a + bytes;
    return wsBase + a;
  };

  int pi = 0;
  const float* x = (const float*)d_in[pi++];
  const float* anchors = (const float*)d_in[n_in - 1];
  const float* w5[5];
  auto P5 = [&]() { for (int j = 0; j < 5; ++j) w5[j] = (const float*)d_in[pi++]; };

  const int B = 2;

  // conv_bn layer driver: pack weights, fold BN, run WMMA conv
  auto conv = [&](const bf16_t* in, bf16_t* out, const bf16_t* res,
                  int Hin, int Win, int Cin, int Cout, int ks, int stride) {
    P5();
    int pad = (ks == 3) ? 1 : 0;
    int Hout = (stride == 2) ? Hin / 2 : Hin;
    int Wout = (stride == 2) ? Win / 2 : Win;
    int K = ks * ks * Cin;
    bf16_t* wp = (bf16_t*)alloc((size_t)Cout * K * sizeof(bf16_t));
    float* sb = (float*)alloc((size_t)Cout * 2 * sizeof(float));
    {
      long long tot = (long long)Cout * K;
      PackP pk{w5[0], wp, K, Cout};
      k_pack_w<<<dim3((unsigned)((tot + 255) / 256)), dim3(256), 0, stream>>>(pk);
    }
    {
      FuseP f{w5[1], w5[2], w5[3], w5[4], sb, Cout};
      k_fuse_bn<<<dim3((unsigned)((Cout + 255) / 256)), dim3(256), 0, stream>>>(f);
    }
    ConvP cp{in, wp, sb, res, out, B, Hin, Win, Cin, Hout, Wout, Cout, ks, stride, pad, K};
    int Npix = B * Hout * Wout;
    k_conv_wmma<<<dim3((unsigned)((Npix + 127) / 128), (unsigned)((Cout + 63) / 64)),
                  dim3(256), 0, stream>>>(cp);
  };

  // ---- Darknet-53 backbone ----
  bf16_t* X416 = (bf16_t*)alloc((size_t)B * 416 * 416 * 32 * sizeof(bf16_t));
  {
    P5();
    C0P p0{x, w5[0], w5[1], w5[2], w5[3], w5[4], X416};
    long long tot = (long long)B * 416 * 416 * 32;
    k_conv0<<<dim3((unsigned)((tot + 255) / 256)), dim3(256), 0, stream>>>(p0);
  }

  static const int SF[5] = {64, 128, 256, 512, 1024};
  static const int SB[5] = {1, 2, 8, 8, 4};
  bf16_t* Xcur = X416; int H = 416; int C = 32;
  bf16_t *x36 = nullptr, *x61 = nullptr;
  for (int si = 0; si < 5; ++si) {
    int f = SF[si]; int Hn = H / 2;
    bf16_t* Xn = (bf16_t*)alloc((size_t)B * Hn * Hn * f * sizeof(bf16_t));
    bf16_t* Hh = (bf16_t*)alloc((size_t)B * Hn * Hn * (f / 2) * sizeof(bf16_t));
    conv(Xcur, Xn, nullptr, H, H, C, f, 3, 2);
    for (int bl = 0; bl < SB[si]; ++bl) {
      conv(Xn, Hh, nullptr, Hn, Hn, f, f / 2, 1, 1);
      conv(Hh, Xn, Xn, Hn, Hn, f / 2, f, 3, 1);   // residual add fused
    }
    if (si == 2) x36 = Xn;
    if (si == 3) x61 = Xn;
    Xcur = Xn; H = Hn; C = f;
  }

  // ---- conv0 tower (13x13) ----
  bf16_t* Ta = (bf16_t*)alloc((size_t)B * 169 * 1024 * sizeof(bf16_t));
  bf16_t* Tb = (bf16_t*)alloc((size_t)B * 169 * 1024 * sizeof(bf16_t));
  {
    const bf16_t* cin = Xcur; int cc = 1024;
    const int fs[5] = {512, 1024, 512, 1024, 512};
    const int ks[5] = {1, 3, 1, 3, 1};
    for (int l = 0; l < 5; ++l) {
      bf16_t* o = (l & 1) ? Tb : Ta;
      conv(cin, o, nullptr, 13, 13, cc, fs[l], ks[l], 1);
      cin = o; cc = fs[l];
    }
  }
  bf16_t* x0 = Ta;  // 512ch

  // head0
  float* HD0 = (float*)alloc((size_t)B * 169 * 255 * sizeof(float));
  conv(x0, Tb, nullptr, 13, 13, 512, 1024, 3, 1);
  {
    const float* hw = (const float*)d_in[pi++];
    const float* hb = (const float*)d_in[pi++];
    HdP hp{Tb, hw, hb, HD0, B * 169, 1024};
    long long tot = (long long)B * 169 * 255;
    k_head<<<dim3((unsigned)((tot + 255) / 256)), dim3(256), 0, stream>>>(hp);
  }

  // pre1 + upsample + concat(x61)
  bf16_t* C26 = (bf16_t*)alloc((size_t)B * 676 * 768 * sizeof(bf16_t));
  conv(x0, Tb, nullptr, 13, 13, 512, 256, 1, 1);
  {
    UpP up{Tb, C26, B, 13, 13, 256, 768, 0};
    long long tot = (long long)B * 676 * 256;
    k_upsample<<<dim3((unsigned)((tot + 255) / 256)), dim3(256), 0, stream>>>(up);
  }
  {
    CpP cp{x61, C26, B, 26, 26, 512, 768, 256};
    long long tot = (long long)B * 676 * 512;
    k_copych<<<dim3((unsigned)((tot + 255) / 256)), dim3(256), 0, stream>>>(cp);
  }

  // conv1 tower (26x26)
  bf16_t* Ua = (bf16_t*)alloc((size_t)B * 676 * 512 * sizeof(bf16_t));
  bf16_t* Ub = (bf16_t*)alloc((size_t)B * 676 * 512 * sizeof(bf16_t));
  {
    const bf16_t* cin = C26; int cc = 768;
    const int fs[5] = {256, 512, 256, 512, 256};
    const int ks[5] = {1, 3, 1, 3, 1};
    for (int l = 0; l < 5; ++l) {
      bf16_t* o = (l & 1) ? Ub : Ua;
      conv(cin, o, nullptr, 26, 26, cc, fs[l], ks[l], 1);
      cin = o; cc = fs[l];
    }
  }
  bf16_t* x1 = Ua;  // 256ch

  // head1
  float* HD1 = (float*)alloc((size_t)B * 676 * 255 * sizeof(float));
  conv(x1, Ub, nullptr, 26, 26, 256, 512, 3, 1);
  {
    const float* hw = (const float*)d_in[pi++];
    const float* hb = (const float*)d_in[pi++];
    HdP hp{Ub, hw, hb, HD1, B * 676, 512};
    long long tot = (long long)B * 676 * 255;
    k_head<<<dim3((unsigned)((tot + 255) / 256)), dim3(256), 0, stream>>>(hp);
  }

  // pre2 + upsample + concat(x36)
  bf16_t* C52 = (bf16_t*)alloc((size_t)B * 2704 * 384 * sizeof(bf16_t));
  conv(x1, Ub, nullptr, 26, 26, 256, 128, 1, 1);
  {
    UpP up{Ub, C52, B, 26, 26, 128, 384, 0};
    long long tot = (long long)B * 2704 * 128;
    k_upsample<<<dim3((unsigned)((tot + 255) / 256)), dim3(256), 0, stream>>>(up);
  }
  {
    CpP cp{x36, C52, B, 52, 52, 256, 384, 128};
    long long tot = (long long)B * 2704 * 256;
    k_copych<<<dim3((unsigned)((tot + 255) / 256)), dim3(256), 0, stream>>>(cp);
  }

  // conv2 tower (52x52)
  bf16_t* Va = (bf16_t*)alloc((size_t)B * 2704 * 256 * sizeof(bf16_t));
  bf16_t* Vb = (bf16_t*)alloc((size_t)B * 2704 * 256 * sizeof(bf16_t));
  {
    const bf16_t* cin = C52; int cc = 384;
    const int fs[5] = {128, 256, 128, 256, 128};
    const int ks[5] = {1, 3, 1, 3, 1};
    for (int l = 0; l < 5; ++l) {
      bf16_t* o = (l & 1) ? Vb : Va;
      conv(cin, o, nullptr, 52, 52, cc, fs[l], ks[l], 1);
      cin = o; cc = fs[l];
    }
  }
  bf16_t* x2 = Va;  // 128ch

  // head2
  float* HD2 = (float*)alloc((size_t)B * 2704 * 255 * sizeof(float));
  conv(x2, Vb, nullptr, 52, 52, 128, 256, 3, 1);
  {
    const float* hw = (const float*)d_in[pi++];
    const float* hb = (const float*)d_in[pi++];
    HdP hp{Vb, hw, hb, HD2, B * 2704, 256};
    long long tot = (long long)B * 2704 * 255;
    k_head<<<dim3((unsigned)((tot + 255) / 256)), dim3(256), 0, stream>>>(hp);
  }

  // ---- decode ----
  float* bbox = (float*)alloc((size_t)B * 10647 * 4 * sizeof(float));
  float* scor = (float*)alloc((size_t)B * 10647 * 80 * sizeof(float));
  auto decode = [&](const float* hd, int G, int rowoff, int abase) {
    DecP dp{hd, anchors, bbox, scor, B, G, rowoff, abase};
    long long tot = (long long)B * G * G * 3;
    k_decode<<<dim3((unsigned)((tot + 255) / 256)), dim3(256), 0, stream>>>(dp);
  };
  decode(HD0, 13, 0, 6);
  decode(HD1, 26, 507, 3);
  decode(HD2, 52, 2535, 0);

  // ---- NMS ----
  float* obx = (float*)alloc((size_t)B * 80 * 256 * 4 * sizeof(float));
  float* osc = (float*)alloc((size_t)B * 80 * 256 * sizeof(float));
  {
    N1P n1{bbox, scor, obx, osc};
    k_nms1<<<dim3(B * 80), dim3(256), 0, stream>>>(n1);
  }
  {
    N2P n2{obx, osc, (float*)d_out};
    k_nms2<<<dim3(B), dim3(256), 0, stream>>>(n2);
  }
}
